// GAT_Net_65163243815810
// MI455X (gfx1250) — matrix operand order repaired and checked by reading, not compiled, not run on hardware
//
#include <hip/hip_runtime.h>

#define HEADS 4

typedef __attribute__((ext_vector_type(16))) __bf16 v16bf;
typedef __attribute__((ext_vector_type(8)))  float  v8f;

// ---------------- helpers ----------------
__device__ __forceinline__ unsigned short f2bf(float f) {
  unsigned u = __float_as_uint(f);
  unsigned r = u + 0x7FFFu + ((u >> 16) & 1u);   // round-to-nearest-even
  return (unsigned short)(r >> 16);
}

__device__ __forceinline__ void atomicMaxF(float* addr, float val) {
  // sign-aware integer punning: monotone under int max (>=0) / uint min (<0)
  if (val >= 0.0f) atomicMax((int*)addr, __float_as_int(val));
  else             atomicMin((unsigned int*)addr, __float_as_uint(val));
}

// ---------------- fill ----------------
__global__ void fill_kernel(float* __restrict__ p, float v, size_t n) {
  size_t i = (size_t)blockIdx.x * blockDim.x + threadIdx.x;
  if (i < n) p[i] = v;
}

// ---------------- f32 -> bf16 (optionally through ELU) ----------------
__global__ void cvt_bf16_kernel(const float* __restrict__ in,
                                unsigned short* __restrict__ out, size_t n) {
  size_t i = (size_t)blockIdx.x * blockDim.x + threadIdx.x;
  if (i < n) out[i] = f2bf(in[i]);
}

__global__ void elu_bf16_kernel(const float* __restrict__ in,
                                unsigned short* __restrict__ out, size_t n) {
  size_t i = (size_t)blockIdx.x * blockDim.x + threadIdx.x;
  if (i < n) {
    float x = in[i];
    float y = (x > 0.0f) ? x : (__expf(x) - 1.0f);
    out[i] = f2bf(y);
  }
}

// ---- repack W [H,K,O] (f32) -> Bt [H*O, K] (bf16, K contiguous per column) ----
__global__ void packW_kernel(const float* __restrict__ W,
                             unsigned short* __restrict__ Bt, int K, int O) {
  int idx = blockIdx.x * blockDim.x + threadIdx.x;
  int total = HEADS * K * O;
  if (idx >= total) return;
  int col = idx / K;
  int k   = idx - col * K;
  int hh  = col / O;
  int o   = col - hh * O;
  Bt[(size_t)col * K + k] = f2bf(W[((size_t)hh * K + k) * O + o]);
}

// ---------------- GEMM: C[M x NT*16] = bf16(A[MxK]) @ bf16(Bt[NT*16 x K])^T + bias ----
// One wave per 16-row block; the wave keeps its A fragment in VGPRs and sweeps all
// NT column tiles, accumulating into NT register tiles (A re-read eliminated).
template<int K, int NT>
__global__ void __launch_bounds__(32)
gemm_wmma_bf16(const unsigned short* __restrict__ A,   // [M x K] row-major bf16
               const unsigned short* __restrict__ Bt,  // [NT*16 x K] bf16 (K contiguous)
               const float* __restrict__ bias,         // [NT*16]
               float* __restrict__ C) {                // [M x NT*16] f32
  constexpr int Ncols = NT * 16;
  const int m0   = blockIdx.x * 16;
  const int lane = threadIdx.x;          // 0..31
  const int half = lane >> 4;            // 0/1
  const int r15  = lane & 15;

  const unsigned short* arow = A + (size_t)(m0 + r15) * K;

  v8f acc[NT];
#pragma unroll
  for (int nt = 0; nt < NT; ++nt) {
    float bv = bias[nt * 16 + r15];
#pragma unroll
    for (int i = 0; i < 8; ++i) acc[nt][i] = bv;
  }

  union Frag { unsigned u[8]; v16bf v; };

#pragma unroll
  for (int kk = 0; kk < K; kk += 32) {
    // A fragment (16x32 bf16): lane half selects K blocks {0..7,16..23} vs {8..15,24..31}
    Frag fa;
#pragma unroll
    for (int i = 0; i < 8; ++i) {
      int kp = (i < 4) ? (2 * i + 8 * half) : (16 + 2 * (i - 4) + 8 * half);
      fa.u[i] = *(const unsigned*)(arow + kk + kp);
    }
#pragma unroll
    for (int nt = 0; nt < NT; ++nt) {
      // B fragment (32x16 bf16): lane half selects K blocks {0..7,16..23} vs {8..15,24..31}
      const unsigned short* bcol = Bt + (size_t)(nt * 16 + r15) * K + kk + 8 * half;
      Frag fb;
#pragma unroll
      for (int j = 0; j < 8; ++j) fb.u[j] = *(const unsigned*)(bcol + ((j < 4) ? 2 * j : 16 + 2 * (j - 4)));
      acc[nt] = __builtin_amdgcn_wmma_f32_16x16x32_bf16(false, fa.v, false, fb.v,
                                                        (short)0, acc[nt], false, false);
    }
  }

  // C/D layout: VGPR r -> rows (r, r+8) for lane halves; col = lane&15
#pragma unroll
  for (int nt = 0; nt < NT; ++nt) {
    float* crow = C + (size_t)(m0 + half * 8) * Ncols + nt * 16 + r15;
#pragma unroll
    for (int r = 0; r < 8; ++r) crow[(size_t)r * Ncols] = acc[nt][r];
  }
}

// ---- per-node attention dots: s_dst[n,h] = Wh[n,h,:].a[h,:O], s_src with a[h,O:] ----
__global__ void sdot_kernel(const float* __restrict__ Wh, const float* __restrict__ a,
                            float* __restrict__ sd, float* __restrict__ ss,
                            int N, int O) {
  int idx = blockIdx.x * blockDim.x + threadIdx.x;
  if (idx >= N * HEADS) return;
  int n = idx >> 2, hh = idx & 3;
  const float* w  = Wh + ((size_t)n * HEADS + hh) * O;
  const float* ad = a + (size_t)hh * 2 * O;
  const float* as = ad + O;
  float x = 0.0f, y = 0.0f;
  for (int o = 0; o < O; ++o) { float t = w[o]; x += t * ad[o]; y += t * as[o]; }
  sd[idx] = x; ss[idx] = y;
}

// ---- pass 1: segment max of leaky_relu attention logits over dst ----
__global__ void edge_max_kernel(const int* __restrict__ src, const int* __restrict__ dst,
                                const float* __restrict__ sd, const float* __restrict__ ss,
                                const float* __restrict__ ab, float* __restrict__ m, int E) {
  int e = blockIdx.x * blockDim.x + threadIdx.x;
  if (e >= E) return;
  int s = src[e], d = dst[e];
#pragma unroll
  for (int hh = 0; hh < HEADS; ++hh) {
    float v = sd[d * HEADS + hh] + ss[s * HEADS + hh] + ab[hh];
    v = (v > 0.0f) ? v : 0.2f * v;
    atomicMaxF(&m[d * HEADS + hh], v);
  }
}

// ---- pass 2: ex = exp(e - m[dst]); denom[dst] += ex; cache ex per edge ----
__global__ void edge_exp_kernel(const int* __restrict__ src, const int* __restrict__ dst,
                                const float* __restrict__ sd, const float* __restrict__ ss,
                                const float* __restrict__ ab, const float* __restrict__ m,
                                float* __restrict__ den, float* __restrict__ ews, int E) {
  int e = blockIdx.x * blockDim.x + threadIdx.x;
  if (e >= E) return;
  int s = src[e], d = dst[e];
#pragma unroll
  for (int hh = 0; hh < HEADS; ++hh) {
    float v = sd[d * HEADS + hh] + ss[s * HEADS + hh] + ab[hh];
    v = (v > 0.0f) ? v : 0.2f * v;
    float ex = __expf(v - m[d * HEADS + hh]);
    ews[(size_t)e * HEADS + hh] = ex;
    atomicAdd(&den[d * HEADS + hh], ex);
  }
}

// ---- pass 3: agg[dst] += (ex/denom[dst]) * Wh[src]; blockDim.x == HEADS*OC ----
__global__ void edge_agg_kernel(const int* __restrict__ src, const int* __restrict__ dst,
                                const float* __restrict__ ews, const float* __restrict__ den,
                                const float* __restrict__ Wh, float* __restrict__ agg,
                                int OC) {
  int e = blockIdx.x;
  int j = threadIdx.x;
  int Wd = blockDim.x;                 // HEADS*OC
  int s = src[e], d = dst[e];
  int hh = j / OC;
  float alpha = ews[(size_t)e * HEADS + hh] / den[d * HEADS + hh];
  float v = Wh[(size_t)s * Wd + j];
  atomicAdd(&agg[(size_t)d * Wd + j], alpha * v);
}

// ---- head-mean -> softmax(16) -> accumulate node-mean into hg[16] ----
__global__ void out_reduce_kernel(const float* __restrict__ agg2,
                                  float* __restrict__ hg, int N, float invN) {
  __shared__ float acc[16];
  int t = threadIdx.x;
  if (t < 16) acc[t] = 0.0f;
  __syncthreads();
  int n = blockIdx.x * blockDim.x + t;
  if (n < N) {
    const float* g = agg2 + (size_t)n * 64;
    float z[16];
    float mx = -1e30f;
#pragma unroll
    for (int c = 0; c < 16; ++c) {
      float v = 0.25f * (g[c] + g[16 + c] + g[32 + c] + g[48 + c]);
      z[c] = v; mx = fmaxf(mx, v);
    }
    float sum = 0.0f;
#pragma unroll
    for (int c = 0; c < 16; ++c) { z[c] = __expf(z[c] - mx); sum += z[c]; }
    float scale = invN / sum;
#pragma unroll
    for (int c = 0; c < 16; ++c) atomicAdd(&acc[c], z[c] * scale);
  }
  __syncthreads();
  if (t < 16) atomicAdd(&hg[t], acc[t]);
}

// ---- final fc: out[c] = hg . fc_w[c,:] + fc_b[c] ----
__global__ void fc_kernel(const float* __restrict__ hg, const float* __restrict__ w,
                          const float* __restrict__ b, float* __restrict__ out) {
  int c = threadIdx.x;
  if (c < 16) {
    float s = b[c];
#pragma unroll
    for (int k = 0; k < 16; ++k) s += hg[k] * w[c * 16 + k];
    out[c] = s;
  }
}

static inline size_t alignUp(size_t x) { return (x + 255) & ~(size_t)255; }
static inline unsigned cdiv(size_t a, unsigned b) { return (unsigned)((a + b - 1) / b); }

extern "C" void kernel_launch(void* const* d_in, const int* in_sizes, int n_in,
                              void* d_out, int out_size, void* d_ws, size_t ws_size,
                              hipStream_t stream) {
  (void)n_in; (void)out_size; (void)ws_size;
  const float* h   = (const float*)d_in[0];
  const int*   src = (const int*)d_in[1];
  const int*   dst = (const int*)d_in[2];
  const float* W1  = (const float*)d_in[3];
  const float* b1  = (const float*)d_in[4];
  const float* a1  = (const float*)d_in[5];
  const float* ab1 = (const float*)d_in[6];
  const float* W2  = (const float*)d_in[7];
  const float* b2  = (const float*)d_in[8];
  const float* a2  = (const float*)d_in[9];
  const float* ab2 = (const float*)d_in[10];
  const float* fcw = (const float*)d_in[11];
  const float* fcb = (const float*)d_in[12];
  float* out = (float*)d_out;

  const size_t N = (size_t)in_sizes[0] / 128;   // 100000 nodes
  const int    E = in_sizes[1];                  // 1600000 edges

  // ---------------- workspace layout (with aliasing after last use) ----------------
  char* ws = (char*)d_ws;
  const size_t szA = alignUp(N * 256 * 4);       // Wh1 | later Wh2 + layer2 scalars
  const size_t szB = alignUp(N * 256 * 4);       // agg1 | later agg2
  const size_t szC = alignUp(N * 256 * 2);       // h1B bf16
  size_t szD = N * 128 * 2;                      // hB bf16 | later ews1/ews2
  if ((size_t)E * HEADS * 4 > szD) szD = (size_t)E * HEADS * 4;
  szD = alignUp(szD);

  char* regA = ws;
  char* regB = regA + szA;
  char* regC = regB + szB;
  char* regD = regC + szC;
  char* regE = regD + szD;

  float* Wh1   = (float*)regA;
  float* Wh2   = (float*)regA;                       // alias: Wh1 dead after layer-1 agg
  float* sdst2 = (float*)(regA + alignUp(N * 64 * 4));
  float* ssrc2 = sdst2 + N * HEADS;
  float* m2    = ssrc2 + N * HEADS;
  float* den2  = m2 + N * HEADS;

  float* agg1 = (float*)regB;
  float* agg2 = (float*)regB;                        // alias: agg1 dead after ELU

  unsigned short* h1B = (unsigned short*)regC;
  unsigned short* hB  = (unsigned short*)regD;       // dead after GEMM1
  float* ews1 = (float*)regD;                        // alias
  float* ews2 = (float*)regD;                        // alias

  char* pE = regE;
  unsigned short* WB1t = (unsigned short*)pE; pE += alignUp(256 * 128 * 2);
  unsigned short* WB2t = (unsigned short*)pE; pE += alignUp(64 * 256 * 2);
  float* sdst1 = (float*)pE; pE += alignUp(N * HEADS * 4);
  float* ssrc1 = (float*)pE; pE += alignUp(N * HEADS * 4);
  float* m1    = (float*)pE; pE += alignUp(N * HEADS * 4);
  float* den1  = (float*)pE; pE += alignUp(N * HEADS * 4);
  float* hg    = (float*)pE; pE += 256;

  const unsigned TB = 256;

  // ---------------- layer 1 ----------------
  fill_kernel<<<cdiv(N * HEADS, TB), TB, 0, stream>>>(m1, -1e30f, N * HEADS);
  fill_kernel<<<cdiv(N * HEADS, TB), TB, 0, stream>>>(den1, 0.0f, N * HEADS);
  fill_kernel<<<cdiv(N * 256, TB), TB, 0, stream>>>(agg1, 0.0f, N * 256);
  fill_kernel<<<1, 16, 0, stream>>>(hg, 0.0f, 16);

  cvt_bf16_kernel<<<cdiv(N * 128, TB), TB, 0, stream>>>(h, hB, N * 128);
  packW_kernel<<<cdiv(HEADS * 128 * 64, TB), TB, 0, stream>>>(W1, WB1t, 128, 64);

  gemm_wmma_bf16<128, 16><<<(unsigned)(N / 16), 32, 0, stream>>>(hB, WB1t, b1, Wh1);

  sdot_kernel<<<cdiv(N * HEADS, TB), TB, 0, stream>>>(Wh1, a1, sdst1, ssrc1, (int)N, 64);
  edge_max_kernel<<<cdiv((size_t)E, TB), TB, 0, stream>>>(src, dst, sdst1, ssrc1, ab1, m1, E);
  edge_exp_kernel<<<cdiv((size_t)E, TB), TB, 0, stream>>>(src, dst, sdst1, ssrc1, ab1, m1, den1, ews1, E);
  edge_agg_kernel<<<(unsigned)E, 256, 0, stream>>>(src, dst, ews1, den1, Wh1, agg1, 64);

  elu_bf16_kernel<<<cdiv(N * 256, TB), TB, 0, stream>>>(agg1, h1B, N * 256);

  // ---------------- layer 2 (init aliased buffers only after their regions are free) --
  fill_kernel<<<cdiv(N * HEADS, TB), TB, 0, stream>>>(m2, -1e30f, N * HEADS);
  fill_kernel<<<cdiv(N * HEADS, TB), TB, 0, stream>>>(den2, 0.0f, N * HEADS);
  fill_kernel<<<cdiv(N * 64, TB), TB, 0, stream>>>(agg2, 0.0f, N * 64);

  packW_kernel<<<cdiv(HEADS * 256 * 16, TB), TB, 0, stream>>>(W2, WB2t, 256, 16);
  gemm_wmma_bf16<256, 4><<<(unsigned)(N / 16), 32, 0, stream>>>(h1B, WB2t, b2, Wh2);

  sdot_kernel<<<cdiv(N * HEADS, TB), TB, 0, stream>>>(Wh2, a2, sdst2, ssrc2, (int)N, 16);
  edge_max_kernel<<<cdiv((size_t)E, TB), TB, 0, stream>>>(src, dst, sdst2, ssrc2, ab2, m2, E);
  edge_exp_kernel<<<cdiv((size_t)E, TB), TB, 0, stream>>>(src, dst, sdst2, ssrc2, ab2, m2, den2, ews2, E);
  edge_agg_kernel<<<(unsigned)E, 64, 0, stream>>>(src, dst, ews2, den2, Wh2, agg2, 16);

  // ---------------- readout ----------------
  out_reduce_kernel<<<cdiv(N, TB), TB, 0, stream>>>(agg2, hg, (int)N, 1.0f / (float)N);
  fc_kernel<<<1, 16, 0, stream>>>(hg, fcw, fcb, out);
}
